// GAT_mol_10754598109978
// MI455X (gfx1250) — matrix operand order repaired and checked by reading, not compiled, not run on hardware
//
#include <hip/hip_runtime.h>
#include <hip/hip_bf16.h>

// ---------------- problem constants (match reference) ----------------
#define NNODES 50000
#define FIN    128
#define NEDGES 800000
#define HEADS  4
#define CCH    64
#define HCH    256      // HEADS*CCH
#define NGRAPH 512
#define NCLS   12
#define NEG_SLOPE 0.2f

typedef __attribute__((ext_vector_type(16))) __bf16         v16bf;
typedef __attribute__((ext_vector_type(16))) unsigned short v16us;
typedef __attribute__((ext_vector_type(8)))  float          v8f;
typedef __attribute__((ext_vector_type(4)))  unsigned int   u32x4;
typedef __attribute__((ext_vector_type(8)))  int            i32x8;
typedef __attribute__((ext_vector_type(4)))  int            i32x4;

#if defined(__has_builtin)
#if __has_builtin(__builtin_amdgcn_tensor_load_to_lds)
#define USE_TDM 1
#endif
#endif

static __device__ __forceinline__ unsigned short f32_to_bf16_rne(float f) {
  unsigned int u = __float_as_uint(f);
  u += 0x7fffu + ((u >> 16) & 1u);           // round-to-nearest-even
  return (unsigned short)(u >> 16);
}

static __device__ __forceinline__ void atomicMaxF32(float* addr, float val) {
  int* ia = (int*)addr;
  int old = *((volatile int*)ia);
  while (__int_as_float(old) < val) {
    int assumed = old;
    old = atomicCAS(ia, assumed, __float_as_int(val));
    if (old == assumed) break;
  }
}

#ifdef USE_TDM
// Issue one 2-D TDM tile load: global [tensor_d1 x tensor_d0] (row stride
// stride_d0 elems, 2-byte elems) -> LDS at lds_byte_addr, tile tile_d1 x tile_d0,
// optional LDS row padding (load side only). Descriptor per CDNA5 ISA 8.3/8.4.
static __device__ __forceinline__ void tdm_load_2d_bf16(
    const unsigned short* gptr, unsigned lds_byte_addr,
    int tensor_d0, int tensor_d1, int stride_d0,
    int tile_d0, int tile_d1,
    int pad_en, int pad_interval, int pad_amount)
{
  unsigned long long ga = (unsigned long long)(size_t)gptr;
  u32x4 g0;
  g0[0] = 1u;                                         // count=1 (user descriptor)
  g0[1] = lds_byte_addr;                              // lds_addr [63:32]
  g0[2] = (unsigned)(ga & 0xffffffffu);               // global_addr lo
  g0[3] = (unsigned)((ga >> 32) & 0x01ffffffu)        // global_addr hi (57-bit)
        | (2u << 30);                                 // type=2 ("image")
  i32x8 g1;
  g1[0] = (1 << 16)                                   // data_size=1 -> 2 bytes
        | (pad_en ? (1 << 20) : 0)
        | (pad_interval << 22) | (pad_amount << 25);  // wg_mask[15:0]=0 (no cluster)
  g1[1] = (tensor_d0 & 0xffff) << 16;                 // tensor_dim0[15:0] @ bits 63:48
  g1[2] = (int)(((unsigned)tensor_d0 >> 16) & 0xffffu)
        | ((tensor_d1 & 0xffff) << 16);               // dim0 hi | tensor_dim1 lo
  g1[3] = (int)(((unsigned)tensor_d1 >> 16) & 0xffffu)
        | ((tile_d0 & 0xffff) << 16);                 // dim1 hi | tile_dim0
  g1[4] = (tile_d1 & 0xffff);                         // tile_dim1 ; tile_dim2=0
  g1[5] = stride_d0;                                  // tensor_dim0_stride lo32
  g1[6] = 0;                                          // stride hi ; dim1 stride (2D: unused)
  g1[7] = 0;
  i32x4 z4 = {0, 0, 0, 0};
#if defined(__clang_major__) && (__clang_major__ >= 23)
  i32x8 z8 = {0, 0, 0, 0, 0, 0, 0, 0};
  __builtin_amdgcn_tensor_load_to_lds(g0, g1, z4, z4, z8, 0);
#else
  __builtin_amdgcn_tensor_load_to_lds(g0, g1, z4, z4, 0);
#endif
}
#endif // USE_TDM

// ---------------- utility kernels ----------------
__global__ void k_cast_bf16(const float* __restrict__ src,
                            unsigned short* __restrict__ dst, long n) {
  long i = (long)blockIdx.x * blockDim.x + threadIdx.x;
  if (i < n) dst[i] = f32_to_bf16_rne(src[i]);
}

__global__ void k_fill_f32(float* __restrict__ p, float v, long n) {
  long i = (long)blockIdx.x * blockDim.x + threadIdx.x;
  if (i < n) p[i] = v;
}

// ---------------- bf16 WMMA GEMM: C[N,M] = A[N,K] x B[K,M] ----------------
// block = 256 threads (8 waves). Block tile 128x64, K-step 32.
// Double-buffered TDM streams LDS slabs (A by wave0, B by wave1 w/ padded rows)
// overlapped with compute; wave w computes rows [16w,16w+16) x cols [0,64):
// one A fragment shared by 4 accumulators (4 WMMAs per K-step per wave).
#define BM 128
#define BN 64
#define BK 32
#define BNP (BN + 16)   // padded B row (elems): 80 shorts = 160 B (32B aligned)

__global__ __launch_bounds__(256) void k_gemm_bf16_wmma(
    const unsigned short* __restrict__ A,   // [Nrows,K] bf16 bits
    const unsigned short* __restrict__ B,   // [K,M]     bf16 bits
    float* __restrict__ C,                  // [Nrows,M] f32
    int Nrows, int K, int M)
{
  __shared__ __attribute__((aligned(32))) unsigned short As[2][BM][BK];   // 2 x 8 KB
  __shared__ __attribute__((aligned(32))) unsigned short Bs[2][BK][BNP];  // 2 x 5 KB

  const int tid  = threadIdx.x;
  const int wave = tid >> 5;
  const int lane = tid & 31;
  const int row0 = blockIdx.x * BM;
  const int col0 = blockIdx.y * BN;
  const int mrow = row0 + (wave << 4);

  v8f acc[4] = {{}, {}, {}, {}};

#ifdef USE_TDM
  // prologue: fill buffer 0
  if (wave == 0) {
    tdm_load_2d_bf16(A + (size_t)row0 * K, (unsigned)(size_t)&As[0][0][0],
                     K, Nrows - row0, K, BK, BM, 0, 0, 0);
  } else if (wave == 1) {
    // pad LDS rows to 80 shorts: interval=32 dw (code 4), amount=8 dw (code 7)
    tdm_load_2d_bf16(B + col0, (unsigned)(size_t)&Bs[0][0][0],
                     M - col0, K, M, BN, BK, 1, 4, 7);
  }
  int cur = 0;
#endif

  for (int k0 = 0; k0 < K; k0 += BK) {
#ifdef USE_TDM
    const int nxt = cur ^ 1;
    const bool has_next = (k0 + BK) < K;
    if (has_next) {
      if (wave == 0) {
        tdm_load_2d_bf16(A + (size_t)row0 * K + (k0 + BK),
                         (unsigned)(size_t)&As[nxt][0][0],
                         K - (k0 + BK), Nrows - row0, K, BK, BM, 0, 0, 0);
      } else if (wave == 1) {
        tdm_load_2d_bf16(B + (size_t)(k0 + BK) * M + col0,
                         (unsigned)(size_t)&Bs[nxt][0][0],
                         M - col0, K - (k0 + BK), M, BN, BK, 1, 4, 7);
      }
    } else if (wave == 2 && k0 + 2 * BK < K) {
      ; // unreachable; keep structure simple
    }
    if (wave == 2 && has_next && k0 + 2 * BK < K) {
      int pr = row0 + (lane << 2); if (pr >= Nrows) pr = Nrows - 1;
      __builtin_prefetch(A + (size_t)pr * K + k0 + 2 * BK, 0, 0);  // global_prefetch_b8
    }
    if (wave < 2) {
      // in-order per-wave TDM: <=1 outstanding ==> current buffer has landed
      if (has_next) __builtin_amdgcn_s_wait_tensorcnt(1);
      else          __builtin_amdgcn_s_wait_tensorcnt(0);
    }
#else
    const int cur = 0;
    // fallback: vector staging through VGPRs
#pragma unroll
    for (int i = 0; i < 2; ++i) {
      int v = tid * 2 + i;
      int r = v >> 2;
      int c = (v & 3) * 8;
      int gr = row0 + r; if (gr >= Nrows) gr = Nrows - 1;
      *(uint4*)&As[0][r][c] = *(const uint4*)(A + (size_t)gr * K + k0 + c);
    }
    {
      int r = tid >> 3;                 // 32 rows, 8 x uint4 per 64-elem row
      int c = (tid & 7) * 8;
      *(uint4*)&Bs[0][r][c] = *(const uint4*)(B + (size_t)(k0 + r) * M + col0 + c);
    }
#endif
    __syncthreads();

    // ---- A fragment (16x32 bf16): lane half selects K-block, pairs per VGPR ----
    v16us au;
    {
      const int m  = lane & 15;
      const int kh = (lane & 16) >> 1;    // 0 for lanes 0-15, 8 for lanes 16-31
#pragma unroll
      for (int i = 0; i < 8; ++i) {
        int k = 2 * i + ((i >= 4) ? 8 : 0) + kh;
        unsigned int d = *(const unsigned int*)&As[cur][(wave << 4) + m][k];
        au[2 * i]     = (unsigned short)(d & 0xffffu);
        au[2 * i + 1] = (unsigned short)(d >> 16);
      }
    }
    v16bf a = __builtin_bit_cast(v16bf, au);

    // ---- B fragments (32x16 bf16 each): lane%16 = N column, 16 K values in A's K order ----
#pragma unroll
    for (int t = 0; t < 4; ++t) {
      v16us bu;
#pragma unroll
      for (int s = 0; s < 16; ++s)
        bu[s] = Bs[cur][((s < 8) ? s : s + 8) + ((lane & 16) >> 1)][t * 16 + (lane & 15)];
      v16bf b  = __builtin_bit_cast(v16bf, bu);
      acc[t] = __builtin_amdgcn_wmma_f32_16x16x32_bf16(false, a, false, b,
                                                       (short)0, acc[t], false, false);
    }
    __syncthreads();
#ifdef USE_TDM
    cur = nxt;
#endif
  }

  // D layout: VGPR r -> (M=r, N=lane) lanes 0-15 ; (M=r+8, N=lane-16) lanes 16-31
  if (mrow < Nrows) {
    const int n  = lane & 15;
    const int mo = (lane >> 4) << 3;
#pragma unroll
    for (int t = 0; t < 4; ++t) {
#pragma unroll
      for (int r = 0; r < 8; ++r) {
        C[(size_t)(mrow + mo + r) * M + col0 + t * 16 + n] = acc[t][r];
      }
    }
  }
}

// ---------------- GATv2 edge kernels (thread per edge*head) ----------------
// Edges e < E0 come from src[]/dst[]; e >= E0 are synthesized self-loops.
__global__ void k_edge_score_max(const float* __restrict__ xl, const float* __restrict__ xr,
                                 const int* __restrict__ src, const int* __restrict__ dst,
                                 const float* __restrict__ att,     // [H][C]
                                 float* __restrict__ score,         // [Etot*H]
                                 float* __restrict__ smax,          // [N*H]
                                 int E0, int Etot, int H, int Cc)
{
  long idx = (long)blockIdx.x * blockDim.x + threadIdx.x;
  if (idx >= (long)Etot * H) return;
  int e = (int)(idx / H), h = (int)(idx % H);
  int s = (e < E0) ? src[e] : (e - E0);
  int d = (e < E0) ? dst[e] : (e - E0);
  const float4* pl = (const float4*)(xl + ((size_t)s * H + h) * Cc);
  const float4* pr = (const float4*)(xr + ((size_t)d * H + h) * Cc);
  const float4* pa = (const float4*)(att + (size_t)h * Cc);
  float acc = 0.f;
#pragma unroll 4
  for (int i = 0; i < Cc / 4; ++i) {
    float4 va = pl[i], vb = pr[i], vw = pa[i];
    float t;
    t = va.x + vb.x; t = (t > 0.f) ? t : NEG_SLOPE * t; acc += vw.x * t;
    t = va.y + vb.y; t = (t > 0.f) ? t : NEG_SLOPE * t; acc += vw.y * t;
    t = va.z + vb.z; t = (t > 0.f) ? t : NEG_SLOPE * t; acc += vw.z * t;
    t = va.w + vb.w; t = (t > 0.f) ? t : NEG_SLOPE * t; acc += vw.w * t;
  }
  score[idx] = acc;
  atomicMaxF32(&smax[(size_t)d * H + h], acc);
}

__global__ void k_edge_exp_sum(const int* __restrict__ dst,
                               float* __restrict__ score,           // in: score, out: exp
                               const float* __restrict__ smax,
                               float* __restrict__ denom,
                               int E0, int Etot, int H)
{
  long idx = (long)blockIdx.x * blockDim.x + threadIdx.x;
  if (idx >= (long)Etot * H) return;
  int e = (int)(idx / H), h = (int)(idx % H);
  int d = (e < E0) ? dst[e] : (e - E0);
  float ex = __expf(score[idx] - smax[(size_t)d * H + h]);
  score[idx] = ex;
  atomicAdd(&denom[(size_t)d * H + h], ex);
}

__global__ void k_edge_agg(const float* __restrict__ xl,
                           const int* __restrict__ src, const int* __restrict__ dst,
                           const float* __restrict__ ex, const float* __restrict__ denom,
                           float* __restrict__ outp,                // [N*H*C]
                           int E0, int Etot, int H, int Cc)
{
  long idx = (long)blockIdx.x * blockDim.x + threadIdx.x;
  if (idx >= (long)Etot * H) return;
  int e = (int)(idx / H), h = (int)(idx % H);
  int s = (e < E0) ? src[e] : (e - E0);
  int d = (e < E0) ? dst[e] : (e - E0);
  float alpha = ex[idx] / (denom[(size_t)d * H + h] + 1e-16f);
  const float* pl = xl + ((size_t)s * H + h) * Cc;
  float* po = outp + ((size_t)d * H + h) * Cc;
#pragma unroll 4
  for (int c = 0; c < Cc; ++c) atomicAdd(&po[c], alpha * pl[c]);
}

// bias (+optional relu) then optional f32 and/or bf16 outputs
__global__ void k_bias_act(const float* __restrict__ in, const float* __restrict__ bias,
                           float* __restrict__ of32, unsigned short* __restrict__ ob16,
                           long n, int Mc, int relu)
{
  long i = (long)blockIdx.x * blockDim.x + threadIdx.x;
  if (i >= n) return;
  float v = in[i] + bias[i % Mc];
  if (relu) v = fmaxf(v, 0.f);
  if (of32) of32[i] = v;
  if (ob16) ob16[i] = f32_to_bf16_rne(v);
}

// ---------------- pooling + head ----------------
__global__ void k_pool_sum(const float* __restrict__ h, const int* __restrict__ batch,
                           float* __restrict__ s, float* __restrict__ cnt)
{
  long i = (long)blockIdx.x * blockDim.x + threadIdx.x;
  if (i >= (long)NNODES * CCH) return;
  int node = (int)(i / CCH), c = (int)(i % CCH);
  int b = batch[node];
  atomicAdd(&s[(size_t)b * CCH + c], h[i]);
  if (c == 0) atomicAdd(&cnt[b], 1.0f);
}

__global__ void k_pool_div(float* __restrict__ s, const float* __restrict__ cnt,
                           unsigned short* __restrict__ gb)
{
  int i = blockIdx.x * blockDim.x + threadIdx.x;
  if (i >= NGRAPH * CCH) return;
  float m = fmaxf(cnt[i / CCH], 1.0f);
  float v = s[i] / m;
  s[i] = v;
  gb[i] = f32_to_bf16_rne(v);
}

__global__ void k_head(const float* __restrict__ g, const float* __restrict__ w,
                       const float* __restrict__ b, float* __restrict__ out)
{
  int r = blockIdx.x * blockDim.x + threadIdx.x;
  if (r >= NGRAPH) return;
  float logit[NCLS];
  float mx = -3e38f;
#pragma unroll
  for (int j = 0; j < NCLS; ++j) {
    float acc = b[j];
    for (int c = 0; c < CCH; ++c) acc += g[(size_t)r * CCH + c] * w[(size_t)c * NCLS + j];
    logit[j] = acc; mx = fmaxf(mx, acc);
  }
  float sum = 0.f;
#pragma unroll
  for (int j = 0; j < NCLS; ++j) { logit[j] = __expf(logit[j] - mx); sum += logit[j]; }
  float inv = 1.f / sum;
#pragma unroll
  for (int j = 0; j < NCLS; ++j) out[(size_t)r * NCLS + j] = logit[j] * inv;
}

// ---------------- host orchestration ----------------
static inline int cdivl(long a, int b) { return (int)((a + b - 1) / b); }

extern "C" void kernel_launch(void* const* d_in, const int* in_sizes, int n_in,
                              void* d_out, int out_size, void* d_ws, size_t ws_size,
                              hipStream_t stream) {
  (void)in_sizes; (void)n_in; (void)out_size; (void)ws_size;
  const float* x    = (const float*)d_in[0];
  const int*   ei   = (const int*)d_in[1];
  const int*   batch= (const int*)d_in[2];
  const float* W1l = (const float*)d_in[3],  *W1r = (const float*)d_in[4];
  const float* a1  = (const float*)d_in[5],  *b1  = (const float*)d_in[6];
  const float* W2l = (const float*)d_in[7],  *W2r = (const float*)d_in[8];
  const float* a2  = (const float*)d_in[9],  *b2  = (const float*)d_in[10];
  const float* W3l = (const float*)d_in[11], *W3r = (const float*)d_in[12];
  const float* a3  = (const float*)d_in[13], *b3  = (const float*)d_in[14];
  const float* l1w = (const float*)d_in[15], *l1b = (const float*)d_in[16];
  const float* l2w = (const float*)d_in[17], *l2b = (const float*)d_in[18];
  const float* l3w = (const float*)d_in[19], *l3b = (const float*)d_in[20];
  float* outp = (float*)d_out;
  const int* srcv = ei;
  const int* dstv = ei + NEDGES;

  // bump allocator over d_ws
  char* cur = (char*)d_ws;
  auto alloc = [&](size_t bytes) -> char* {
    char* p = cur; cur += (bytes + 255) & ~(size_t)255; return p;
  };
  unsigned short* xb  = (unsigned short*)alloc((size_t)NNODES * FIN * 2);
  unsigned short* hb  = (unsigned short*)alloc((size_t)NNODES * HCH * 2);
  float* xl   = (float*)alloc((size_t)NNODES * HCH * 4);
  float* xr   = (float*)alloc((size_t)NNODES * HCH * 4);
  float* agg  = (float*)alloc((size_t)NNODES * HCH * 4);
  float* ebuf = (float*)alloc((size_t)(NEDGES + NNODES) * HEADS * 4);
  float* smax = (float*)alloc((size_t)NNODES * HEADS * 4);
  float* den  = (float*)alloc((size_t)NNODES * HEADS * 4);
  unsigned short* w1lb = (unsigned short*)alloc((size_t)FIN * HCH * 2);
  unsigned short* w1rb = (unsigned short*)alloc((size_t)FIN * HCH * 2);
  unsigned short* w2lb = (unsigned short*)alloc((size_t)HCH * HCH * 2);
  unsigned short* w2rb = (unsigned short*)alloc((size_t)HCH * HCH * 2);
  unsigned short* w3lb = (unsigned short*)alloc((size_t)HCH * CCH * 2);
  unsigned short* w3rb = (unsigned short*)alloc((size_t)HCH * CCH * 2);
  unsigned short* l1wb = (unsigned short*)alloc((size_t)CCH * CCH * 2);
  unsigned short* l2wb = (unsigned short*)alloc((size_t)CCH * CCH * 2);
  float* pool_s = (float*)alloc((size_t)NGRAPH * CCH * 4);
  float* pool_c = (float*)alloc((size_t)NGRAPH * 4);
  unsigned short* gb = (unsigned short*)alloc((size_t)NGRAPH * CCH * 2);
  float* g1 = (float*)alloc((size_t)NGRAPH * CCH * 4);
  float* g2 = (float*)alloc((size_t)NGRAPH * CCH * 4);

  auto cast = [&](const float* s, unsigned short* d, long n) {
    k_cast_bf16<<<cdivl(n, 256), 256, 0, stream>>>(s, d, n);
  };
  auto fill = [&](float* p, float v, long n) {
    k_fill_f32<<<cdivl(n, 256), 256, 0, stream>>>(p, v, n);
  };
  auto gemm = [&](const unsigned short* A, const unsigned short* B, float* C,
                  int Nr, int K, int M) {
    dim3 grid(cdivl(Nr, BM), M / BN);
    k_gemm_bf16_wmma<<<grid, 256, 0, stream>>>(A, B, C, Nr, K, M);
  };
  auto conv = [&](const float* att, int E0, int Etot, int H) {
    fill(smax, -3.0e38f, (long)NNODES * H);
    fill(den, 0.f, (long)NNODES * H);
    fill(agg, 0.f, (long)NNODES * H * CCH);
    long tot = (long)Etot * H;
    k_edge_score_max<<<cdivl(tot, 256), 256, 0, stream>>>(xl, xr, srcv, dstv, att,
                                                          ebuf, smax, E0, Etot, H, CCH);
    k_edge_exp_sum<<<cdivl(tot, 256), 256, 0, stream>>>(dstv, ebuf, smax, den, E0, Etot, H);
    k_edge_agg<<<cdivl(tot, 256), 256, 0, stream>>>(xl, srcv, dstv, ebuf, den, agg,
                                                    E0, Etot, H, CCH);
  };
  auto biasact = [&](const float* in, const float* bias, float* of32,
                     unsigned short* ob16, long n, int Mc, int relu) {
    k_bias_act<<<cdivl(n, 256), 256, 0, stream>>>(in, bias, of32, ob16, n, Mc, relu);
  };

  // ---- precision casts ----
  cast(x, xb, (long)NNODES * FIN);
  cast(W1l, w1lb, (long)FIN * HCH);  cast(W1r, w1rb, (long)FIN * HCH);
  cast(W2l, w2lb, (long)HCH * HCH);  cast(W2r, w2rb, (long)HCH * HCH);
  cast(W3l, w3lb, (long)HCH * CCH);  cast(W3r, w3rb, (long)HCH * CCH);
  cast(l1w, l1wb, (long)CCH * CCH);  cast(l2w, l2wb, (long)CCH * CCH);

  // ---- conv1 (no self loops, H=4) ----
  gemm(xb, w1lb, xl, NNODES, FIN, HCH);
  gemm(xb, w1rb, xr, NNODES, FIN, HCH);
  conv(a1, NEDGES, NEDGES, HEADS);
  biasact(agg, b1, nullptr, hb, (long)NNODES * HCH, HCH, 1);

  // ---- conv2 (self loops, H=4) ----
  gemm(hb, w2lb, xl, NNODES, HCH, HCH);
  gemm(hb, w2rb, xr, NNODES, HCH, HCH);
  conv(a2, NEDGES, NEDGES + NNODES, HEADS);
  biasact(agg, b2, nullptr, hb, (long)NNODES * HCH, HCH, 1);

  // ---- conv3 (self loops, H=1) ----
  gemm(hb, w3lb, xl, NNODES, HCH, CCH);
  gemm(hb, w3rb, xr, NNODES, HCH, CCH);
  conv(a3, NEDGES, NEDGES + NNODES, 1);
  float* h3 = xr;   // xr free after the score kernel; reuse for f32 conv3 output
  biasact(agg, b3, h3, nullptr, (long)NNODES * CCH, CCH, 0);

  // ---- global mean pool ----
  fill(pool_s, 0.f, (long)NGRAPH * CCH);
  fill(pool_c, 0.f, NGRAPH);
  k_pool_sum<<<cdivl((long)NNODES * CCH, 256), 256, 0, stream>>>(h3, batch, pool_s, pool_c);
  k_pool_div<<<cdivl((long)NGRAPH * CCH, 256), 256, 0, stream>>>(pool_s, pool_c, gb);

  // ---- MLP head ----
  gemm(gb, l1wb, g1, NGRAPH, CCH, CCH);
  biasact(g1, l1b, nullptr, gb, (long)NGRAPH * CCH, CCH, 1);
  gemm(gb, l2wb, g1, NGRAPH, CCH, CCH);
  biasact(g1, l2b, g2, nullptr, (long)NGRAPH * CCH, CCH, 1);
  k_head<<<cdivl(NGRAPH, 128), 128, 0, stream>>>(g2, l3w, l3b, outp);
}
